// EncoderRNN_78778290143391
// MI455X (gfx1250) — compile-verified, matrix operand order checked by reading, not run on hardware
//
#include <hip/hip_runtime.h>
#include <hip/hip_bf16.h>
#include <stdint.h>

// ---------------------------------------------------------------------------
// EncoderRNN on MI455X (gfx1250): 2-layer "bi" GRU (both dirs forward) + pool+FC
//   - Input-gate GEMMs: 128x128 LDS-tiled bf16 WMMA, double-buffered
//     global_load_async_to_lds_b128 staging (ASYNCcnt-paced pipeline).
//   - Recurrent scan: 8 blocks = (4 batch-tiles x 2 dirs), h tile in LDS,
//     per-step 16x1536x512 GEMM on v_wmma_f32_16x16x32_bf16, gates fused in
//     registers (each wave owns matching r/z/n column ranges).
// ---------------------------------------------------------------------------

#define Bc   64
#define Tc   448
#define DIN  256
#define Hc   512
#define H3   1536
#define BT   (Bc * Tc)
#define LDH  520            // padded bf16 row stride for h tile (bank spread)
#define LDT  40             // padded bf16 row stride for GEMM LDS tiles

typedef __attribute__((ext_vector_type(16))) __bf16     v16bf;
typedef __attribute__((ext_vector_type(8)))  float      v8f;
typedef __attribute__((ext_vector_type(4)))  unsigned   u32x4;

union V16U { v16bf v; u32x4 q[2]; };

__device__ __forceinline__ unsigned short f32_to_bf16(float f) {
    unsigned x = __float_as_uint(f);
    return (unsigned short)((x + 0x7FFFu + ((x >> 16) & 1u)) >> 16);
}
__device__ __forceinline__ float bf16_to_f32(unsigned short u) {
    return __uint_as_float(((unsigned)u) << 16);
}

// Load one 16x32 bf16 WMMA operand tile (16 rows x 32 K).
// ISA layout: lane L holds row (L&15), K-chunks [ko,ko+8) and [ko+16,ko+24),
// ko = (L>>4)*8  -> two 16-byte loads.
__device__ __forceinline__ v16bf load_tile(const unsigned short* __restrict__ base,
                                           int ld) {
    const int lane = threadIdx.x & 31;
    const int r    = lane & 15;
    const int ko   = (lane >> 4) << 3;
    const unsigned short* p = base + (size_t)r * ld + ko;
    V16U u;
    u.q[0] = *(const u32x4*)(p);
    u.q[1] = *(const u32x4*)(p + 16);
    return u.v;
}

__device__ __forceinline__ v8f wmma_bf16(v16bf a, v16bf b, v8f c) {
    return __builtin_amdgcn_wmma_f32_16x16x32_bf16(false, a, false, b,
                                                   (short)0, c, false, false);
}

// CDNA5 async copy: 16 bytes global -> LDS per lane, tracked by ASYNCcnt.
// GVS mode: mem_addr = SADDR(64b base) + VADDR(32b byte offset).
__device__ __forceinline__ void async_cp16(unsigned lds_byte,
                                           const unsigned short* gbase,
                                           unsigned gofs_bytes) {
    asm volatile("global_load_async_to_lds_b128 %0, %1, %2"
                 :: "v"(lds_byte), "v"(gofs_bytes), "s"(gbase)
                 : "memory");
}
#define WAIT_ASYNC(N) asm volatile("s_wait_asynccnt " #N ::: "memory")

// --------------------------- fp32 -> bf16 convert ---------------------------
__global__ void cvt_f32_bf16(const float* __restrict__ src,
                             unsigned short* __restrict__ dst, long long n) {
    long long i = (long long)blockIdx.x * blockDim.x + threadIdx.x;
    if (i < n) dst[i] = f32_to_bf16(src[i]);
}

// ------------------- C[M,N] = A[M,K] @ W[N,K]^T + bias[N] -------------------
// 256 threads = 8 waves (2x4), 128x128 tile, K-step 32, double-buffered
// async-to-LDS staging. Requires M%128==0, N%128==0, K%32==0 (true here).
__global__ void gemm_bias_wmma(const unsigned short* __restrict__ A,
                               const unsigned short* __restrict__ W,
                               const float* __restrict__ bias,
                               float* __restrict__ C, int M, int N, int K) {
    __shared__ unsigned short Abuf[2][128 * LDT];
    __shared__ unsigned short Bbuf[2][128 * LDT];

    const int tid  = threadIdx.x;
    const int lane = tid & 31;
    const int col  = lane & 15;
    const int mh   = lane >> 4;
    const int wv   = tid >> 5;
    const int wr   = wv & 1;        // 0..1 : 64-row half
    const int wc   = wv >> 1;       // 0..3 : 32-col quarter

    const int Nt = N >> 7;
    const int m0 = (blockIdx.x / Nt) << 7;
    const int n0 = (blockIdx.x % Nt) << 7;

    // Each stage = 4 async instructions per wave (2 for A, 2 for B).
    auto stage = [&](int b, int k0) {
#pragma unroll
        for (int i = 0; i < 2; ++i) {
            const int c   = tid + i * 256;      // 0..511 chunk id
            const int row = c >> 2;
            const int cc  = c & 3;              // 16B chunk within 64B row
            async_cp16((unsigned)(uintptr_t)&Abuf[b][row * LDT + cc * 8],
                       A, (unsigned)(((size_t)(m0 + row) * K + k0 + cc * 8) * 2));
        }
#pragma unroll
        for (int i = 0; i < 2; ++i) {
            const int c   = tid + i * 256;
            const int row = c >> 2;
            const int cc  = c & 3;
            async_cp16((unsigned)(uintptr_t)&Bbuf[b][row * LDT + cc * 8],
                       W, (unsigned)(((size_t)(n0 + row) * K + k0 + cc * 8) * 2));
        }
    };

    v8f acc[4][2] = {};
    const int KT = K >> 5;

    stage(0, 0);
    for (int kt = 0; kt < KT; ++kt) {
        if (kt + 1 < KT) {
            stage((kt + 1) & 1, (kt + 1) << 5);
            WAIT_ASYNC(4);      // drain older stage; prefetch stays in flight
        } else {
            WAIT_ASYNC(0);
        }
        __syncthreads();        // stage kt visible to all waves

        const unsigned short* Ab = Abuf[kt & 1];
        const unsigned short* Bb = Bbuf[kt & 1];
        v16bf bfr[2];
#pragma unroll
        for (int ni = 0; ni < 2; ++ni)
            bfr[ni] = load_tile(Bb + (wc * 32 + ni * 16) * LDT, LDT);
#pragma unroll
        for (int mi = 0; mi < 4; ++mi) {
            v16bf a = load_tile(Ab + (wr * 64 + mi * 16) * LDT, LDT);
#pragma unroll
            for (int ni = 0; ni < 2; ++ni)
                acc[mi][ni] = wmma_bf16(a, bfr[ni], acc[mi][ni]);
        }
        __syncthreads();        // reads done before next prefetch overwrites
    }

#pragma unroll
    for (int mi = 0; mi < 4; ++mi) {
#pragma unroll
        for (int ni = 0; ni < 2; ++ni) {
            const int n = n0 + wc * 32 + ni * 16 + col;
            const float bb = bias[n];
#pragma unroll
            for (int i = 0; i < 8; ++i) {
                const int m = m0 + wr * 64 + mi * 16 + mh * 8 + i;
                C[(size_t)m * N + n] = acc[mi][ni][i] + bb;
            }
        }
    }
}

// ------------------------------- GRU scan -----------------------------------
// grid = (4 batch-tiles, 2 dirs), block = 256 (8 waves).
// Wave w owns columns j in [w*64, w*64+64) of ALL THREE gate blocks
// (r at j, z at 512+j, n at 1024+j) -> gates fuse in registers.
__global__ void gru_scan(const float* __restrict__ gx_all,          // [2][B,T,3H]
                         const unsigned short* __restrict__ whh_all,// [2][3H,H] bf16
                         const float* __restrict__ bhh_all,         // [2][3H]
                         unsigned short* __restrict__ outb,         // [B,T,2H] bf16 or null
                         float* __restrict__ outf) {                // [B,T,2H] f32 or null
    __shared__ unsigned short hs[16 * LDH];     // h tile, bf16

    const int dir  = blockIdx.y;
    const int b0   = blockIdx.x * 16;
    const float* gx            = gx_all + (size_t)dir * BT * H3;
    const unsigned short* whh  = whh_all + (size_t)dir * H3 * Hc;
    const float* bhh           = bhh_all + dir * H3;

    const int wave = threadIdx.x >> 5;
    const int lane = threadIdx.x & 31;
    const int col  = lane & 15;
    const int mh   = lane >> 4;

    for (int i = threadIdx.x; i < 16 * LDH; i += blockDim.x) hs[i] = 0;
    __syncthreads();

    for (int t = 0; t < Tc; ++t) {
        // ---- phase 1: gh = h @ w_hh^T for this wave's 12 tiles (3 gates x 4)
        v8f acc[3][4] = {};
        for (int k0 = 0; k0 < Hc; k0 += 32) {
            v16bf a = load_tile(hs + k0, LDH);
#pragma unroll
            for (int g = 0; g < 3; ++g) {
#pragma unroll
                for (int jt = 0; jt < 4; ++jt) {
                    const int n0 = g * Hc + wave * 64 + jt * 16;
                    v16bf b = load_tile(whh + (size_t)n0 * Hc + k0, Hc);
                    acc[g][jt] = wmma_bf16(a, b, acc[g][jt]);
                }
            }
        }
        __syncthreads();   // all reads of hs complete before we overwrite it

        // ---- phase 2: gates + h update, all in registers
#pragma unroll
        for (int jt = 0; jt < 4; ++jt) {
            const int j  = wave * 64 + jt * 16 + col;
            const float br = bhh[j], bz = bhh[Hc + j], bn = bhh[2 * Hc + j];
#pragma unroll
            for (int i = 0; i < 8; ++i) {
                const int row = mh * 8 + i;
                const float* gxr = gx + ((size_t)(b0 + row) * Tc + t) * H3;
                const float xr = gxr[j];
                const float xz = gxr[Hc + j];
                const float xn = gxr[2 * Hc + j];
                const float hr = acc[0][jt][i] + br;
                const float hz = acc[1][jt][i] + bz;
                const float hn = acc[2][jt][i] + bn;
                const float hp = bf16_to_f32(hs[row * LDH + j]);
                const float r  = 1.0f / (1.0f + __expf(-(xr + hr)));
                const float z  = 1.0f / (1.0f + __expf(-(xz + hz)));
                const float nn = tanhf(xn + r * hn);
                const float h  = (1.0f - z) * nn + z * hp;
                hs[row * LDH + j] = f32_to_bf16(h);
                const size_t o = (((size_t)(b0 + row)) * Tc + t) * (2 * Hc)
                               + (size_t)dir * Hc + j;
                if (outb) outb[o] = f32_to_bf16(h);
                else      outf[o] = h;
            }
        }
        __syncthreads();   // h fully updated before next step's GEMM reads
    }
}

// ---------------- pool (T 448->224 bins of 2, mean over 2H) + FC ------------
__global__ void pool_fc(const float* __restrict__ h2,   // [B,T,2H] f32
                        const float* __restrict__ fcw,  // [224]
                        const float* __restrict__ fcb,  // [1]
                        float* __restrict__ out) {      // [B]
    __shared__ float red[256];
    const int b = blockIdx.x;
    float acc = 0.0f;
    for (int idx = threadIdx.x; idx < Tc * 2 * Hc; idx += blockDim.x) {
        const int t = idx >> 10;              // / (2H = 1024)
        acc += h2[(size_t)b * Tc * 1024 + idx] * fcw[t >> 1];
    }
    red[threadIdx.x] = acc;
    __syncthreads();
    for (int s = 128; s > 0; s >>= 1) {
        if ((int)threadIdx.x < s) red[threadIdx.x] += red[threadIdx.x + s];
        __syncthreads();
    }
    if (threadIdx.x == 0) out[b] = red[0] * (1.0f / 2048.0f) + fcb[0];
}

// ---------------------------------------------------------------------------
extern "C" void kernel_launch(void* const* d_in, const int* in_sizes, int n_in,
                              void* d_out, int out_size, void* d_ws, size_t ws_size,
                              hipStream_t stream) {
    const float* x       = (const float*)d_in[0];
    // d_in[1] = seq_len (compile-time Tc here)
    const float* w_ih_l0 = (const float*)d_in[2];
    const float* w_hh_l0 = (const float*)d_in[3];
    const float* b_ih_l0 = (const float*)d_in[4];
    const float* b_hh_l0 = (const float*)d_in[5];
    const float* w_ih_l1 = (const float*)d_in[6];
    const float* w_hh_l1 = (const float*)d_in[7];
    const float* b_ih_l1 = (const float*)d_in[8];
    const float* b_hh_l1 = (const float*)d_in[9];
    const float* fc_w    = (const float*)d_in[10];
    const float* fc_b    = (const float*)d_in[11];
    float* out           = (float*)d_out;

    // -------- workspace carve (256B aligned) --------
    char* ws = (char*)d_ws;
    size_t off = 0;
    auto carve = [&](size_t bytes) -> char* {
        char* p = ws + off;
        off = (off + bytes + 255) & ~(size_t)255;
        return p;
    };
    unsigned short* xb    = (unsigned short*)carve((size_t)BT * DIN * 2);
    unsigned short* wih0b = (unsigned short*)carve((size_t)2 * H3 * DIN * 2);
    unsigned short* whh0b = (unsigned short*)carve((size_t)2 * H3 * Hc * 2);
    unsigned short* wih1b = (unsigned short*)carve((size_t)2 * H3 * (2 * Hc) * 2);
    unsigned short* whh1b = (unsigned short*)carve((size_t)2 * H3 * Hc * 2);
    float*          gx    = (float*)carve((size_t)2 * BT * H3 * 4);
    unsigned short* h1b   = (unsigned short*)carve((size_t)BT * 2 * Hc * 2);
    float*          h2f   = (float*)carve((size_t)BT * 2 * Hc * 4);
    (void)ws_size; (void)in_sizes; (void)n_in; (void)out_size;

    // -------- convert activations/weights to bf16 --------
    auto cvt = [&](const float* s, unsigned short* d, long long n) {
        cvt_f32_bf16<<<(unsigned)((n + 255) / 256), 256, 0, stream>>>(s, d, n);
    };
    cvt(x,       xb,    (long long)BT * DIN);
    cvt(w_ih_l0, wih0b, (long long)2 * H3 * DIN);
    cvt(w_hh_l0, whh0b, (long long)2 * H3 * Hc);
    cvt(w_ih_l1, wih1b, (long long)2 * H3 * 2 * Hc);
    cvt(w_hh_l1, whh1b, (long long)2 * H3 * Hc);

    const int gblks = (BT / 128) * (H3 / 128);   // 224 * 12 = 2688 tiles

    // -------- layer 0: gx = x @ w_ih^T + b_ih, then scan --------
    for (int d = 0; d < 2; ++d) {
        gemm_bias_wmma<<<gblks, 256, 0, stream>>>(
            xb, wih0b + (size_t)d * H3 * DIN, b_ih_l0 + (size_t)d * H3,
            gx + (size_t)d * BT * H3, BT, H3, DIN);
    }
    gru_scan<<<dim3(4, 2), 256, 0, stream>>>(gx, whh0b, b_hh_l0, h1b, nullptr);

    // -------- layer 1: gx = h1 @ w_ih^T + b_ih, then scan --------
    for (int d = 0; d < 2; ++d) {
        gemm_bias_wmma<<<gblks, 256, 0, stream>>>(
            h1b, wih1b + (size_t)d * H3 * 2 * Hc, b_ih_l1 + (size_t)d * H3,
            gx + (size_t)d * BT * H3, BT, H3, 2 * Hc);
    }
    gru_scan<<<dim3(4, 2), 256, 0, stream>>>(gx, whh1b, b_hh_l1, nullptr, h2f);

    // -------- pool + FC --------
    pool_fc<<<Bc, 256, 0, stream>>>(h2f, fc_w, fc_b, out);
}
